// MoE_70136815943761
// MI455X (gfx1250) — matrix operand order repaired and checked by reading, not compiled, not run on hardware
//
#include <hip/hip_runtime.h>

#define DIM    1024
#define INTERN 1024
#define NE     8
#define T_TOK  2048

typedef __bf16        v16bf __attribute__((ext_vector_type(16)));
typedef float         v8f   __attribute__((ext_vector_type(8)));
typedef unsigned int  v4u   __attribute__((ext_vector_type(4)));
typedef unsigned int  v8u   __attribute__((ext_vector_type(8)));

__device__ __forceinline__ unsigned short f2bf(float f) {
    unsigned u = __float_as_uint(f);
    u += 0x7fffu + ((u >> 16) & 1u);          // round-to-nearest-even
    return (unsigned short)(u >> 16);
}

__device__ __forceinline__ v16bf load_frag(const unsigned short* p0,
                                           const unsigned short* p1) {
    v8u t;
    t.lo = *(const v4u*)p0;
    t.hi = *(const v4u*)p1;
    return __builtin_bit_cast(v16bf, t);
}

// CDNA5 async copy: 16B per lane, global -> LDS, tracked by ASYNCcnt.
__device__ __forceinline__ void async_b128(unsigned lds_off, const void* gptr) {
    unsigned long long ga = (unsigned long long)gptr;
    asm volatile("global_load_async_to_lds_b128 %0, %1, off"
                 :: "v"(lds_off), "v"(ga) : "memory");
}
__device__ __forceinline__ unsigned lds_off32(const void* p) {
    return (unsigned)(unsigned long long)p;   // low 32 bits of flat LDS addr = LDS offset
}
#define WAIT_ASYNC(n) asm volatile("s_wait_asynccnt " #n ::: "memory")

// ---------------------------------------------------------------------------
// One-time per launch: fp32 -> bf16 conversions.
// ---------------------------------------------------------------------------
__launch_bounds__(256)
__global__ void convert_x(const float* __restrict__ src,
                          unsigned short* __restrict__ dst) {
    long i4 = (long)blockIdx.x * 256 + threadIdx.x;     // float4 index
    float4 v = *(const float4*)(src + i4 * 4);
    uint2 p;
    p.x = (unsigned)f2bf(v.x) | ((unsigned)f2bf(v.y) << 16);
    p.y = (unsigned)f2bf(v.z) | ((unsigned)f2bf(v.w) << 16);
    *(uint2*)(dst + i4 * 4) = p;
}

// src[m][k][n] fp32 (1024x1024 per matrix) -> dst[m][n][k] bf16 (B^T, col-major)
__launch_bounds__(256)
__global__ void convert_transpose(const float* __restrict__ src,
                                  unsigned short* __restrict__ dst) {
    __shared__ float tile[32][33];
    int m  = blockIdx.z;
    int nb = blockIdx.x * 32;
    int kb = blockIdx.y * 32;
    int tx = threadIdx.x & 31;
    int ty = threadIdx.x >> 5;                 // 0..7
    const float* s = src + ((long)m << 20) + (long)kb * 1024 + nb;
#pragma unroll
    for (int r = 0; r < 4; ++r)
        tile[ty + r * 8][tx] = s[(long)(ty + r * 8) * 1024 + tx];
    __syncthreads();
    unsigned short* d = dst + ((long)m << 20) + (long)nb * 1024 + kb;
#pragma unroll
    for (int r = 0; r < 4; ++r)
        d[(long)(ty + r * 8) * 1024 + tx] = f2bf(tile[tx][ty + r * 8]);
}

// ---------------------------------------------------------------------------
// Router: logits = x @ gate_w, softmax, top-2; per-expert gather lists.
// ---------------------------------------------------------------------------
__launch_bounds__(256)
__global__ void moe_router(const float* __restrict__ x,
                           const float* __restrict__ gw,
                           float* __restrict__ top_wt,
                           int*   __restrict__ cnt,
                           int*   __restrict__ lists) {
    int t = blockIdx.x * blockDim.x + threadIdx.x;
    if (t >= T_TOK) return;

    float acc[NE];
#pragma unroll
    for (int e = 0; e < NE; ++e) acc[e] = 0.f;
    const float* xr = x + (long)t * DIM;
    for (int d = 0; d < DIM; d += 4) {
        float4 xv = *(const float4*)(xr + d);
        const float* g = gw + (long)d * NE;
#pragma unroll
        for (int e = 0; e < NE; ++e)
            acc[e] += xv.x * g[e] + xv.y * g[NE + e] +
                      xv.z * g[2 * NE + e] + xv.w * g[3 * NE + e];
    }
    float mx = acc[0];
#pragma unroll
    for (int e = 1; e < NE; ++e) mx = fmaxf(mx, acc[e]);
    float p[NE], sum = 0.f;
#pragma unroll
    for (int e = 0; e < NE; ++e) { p[e] = __expf(acc[e] - mx); sum += p[e]; }

    int i0 = 0; float v0 = p[0];
#pragma unroll
    for (int e = 1; e < NE; ++e) if (p[e] > v0) { v0 = p[e]; i0 = e; }
    int i1 = -1; float v1 = -1.f;
#pragma unroll
    for (int e = 0; e < NE; ++e) if (e != i0 && p[e] > v1) { v1 = p[e]; i1 = e; }

    float inv = 1.0f / sum;
    top_wt[t * 2 + 0] = v0 * inv;
    top_wt[t * 2 + 1] = v1 * inv;
    int p0 = atomicAdd(&cnt[i0], 1); lists[i0 * T_TOK + p0] = t * 2;
    int p1 = atomicAdd(&cnt[i1], 1); lists[i1 * T_TOK + p1] = t * 2 + 1;
}

// ---------------------------------------------------------------------------
// Pass 1: H[code] = silu(X@B1)*(X@B3). X bf16 [*,1024]; B bf16 col-major [N][K].
// Block tile 64x128, K step 32, double-buffered async LDS staging.
// ---------------------------------------------------------------------------
__launch_bounds__(256)
__global__ void moe_gemm_swiglu(const unsigned short* __restrict__ Xb,
                                const unsigned short* __restrict__ B1t,
                                const unsigned short* __restrict__ B3t,
                                unsigned short* __restrict__ Hout,
                                const int* __restrict__ rows_all,
                                const int* __restrict__ cnt,
                                int fixedM, int shiftA, long estride) {
    __shared__ __align__(16) unsigned short As[2][64 * 40];
    __shared__ __align__(16) unsigned short Bs1[2][128 * 40];
    __shared__ __align__(16) unsigned short Bs3[2][128 * 40];
    __shared__ int s_code[64];
    __shared__ int s_arow[64];

    int bx = blockIdx.x;
    int nt = bx & 7;
    int mt = (bx >> 3) & 31;
    int e  = bx >> 8;
    int M  = cnt ? cnt[e] : fixedM;
    if (mt * 64 >= M) return;

    const int* rows = rows_all ? (rows_all + (long)e * T_TOK) : nullptr;
    const unsigned short* B1 = B1t + (long)e * estride;
    const unsigned short* B3 = B3t + (long)e * estride;

    int tid = threadIdx.x;
    if (tid < 64) {
        int gr = mt * 64 + tid;
        int code = (gr < M) ? (rows ? rows[gr] : gr) : -1;
        s_code[tid] = code;
        s_arow[tid] = (code >= 0) ? (code >> shiftA) : 0;
    }
    __syncthreads();

    int lane = tid & 31, wave = tid >> 5;
    int wm = wave & 3, wn = wave >> 2;
    int lrow = lane & 15, lhalf = lane >> 4;
    int n0 = nt * 128;

    int a_row = tid >> 2, a_seg = tid & 3;                 // A: 1 xfer/thread
    // staging: 5 async b128 per thread per slab (A + 2xB1 + 2xB3)
    auto stage = [&](int k0, int buf) {
        async_b128(lds_off32(&As[buf][a_row * 40 + a_seg * 8]),
                   Xb + (long)s_arow[a_row] * DIM + k0 + a_seg * 8);
#pragma unroll
        for (int i = 0; i < 2; ++i) {
            int q = tid + i * 256;
            int col = q >> 2, seg = q & 3;
            long go = (long)(n0 + col) * DIM + k0 + seg * 8;
            async_b128(lds_off32(&Bs1[buf][col * 40 + seg * 8]), B1 + go);
            async_b128(lds_off32(&Bs3[buf][col * 40 + seg * 8]), B3 + go);
        }
    };

    v8f zero8 = {0.f, 0.f, 0.f, 0.f, 0.f, 0.f, 0.f, 0.f};
    v8f acc1[4], acc3[4];
#pragma unroll
    for (int j = 0; j < 4; ++j) { acc1[j] = zero8; acc3[j] = zero8; }

    stage(0, 0);
    for (int ks = 0; ks < DIM / 32; ++ks) {
        int cur = ks & 1;
        if (ks + 1 < DIM / 32) {
            stage((ks + 1) * 32, cur ^ 1);
            WAIT_ASYNC(0x5);                 // previous slab complete (in-order)
        } else {
            WAIT_ASYNC(0x0);
        }
        __syncthreads();

        // Load all fragments first ...
        const unsigned short* ap = &As[cur][(wm * 16 + lrow) * 40 + lhalf * 16];
        v16bf af = load_frag(ap, ap + 8);
        v16bf f1[4], f3[4];
#pragma unroll
        for (int j = 0; j < 4; ++j) {
            const unsigned short* bp1 = &Bs1[cur][(wn * 64 + j * 16 + lrow) * 40 + lhalf * 16];
            f1[j] = load_frag(bp1, bp1 + 8);
            const unsigned short* bp3 = &Bs3[cur][(wn * 64 + j * 16 + lrow) * 40 + lhalf * 16];
            f3[j] = load_frag(bp3, bp3 + 8);
        }
        // ... then fire the WMMA burst behind a single dscnt wait.
        __builtin_amdgcn_sched_barrier(0);
#pragma unroll
        for (int j = 0; j < 4; ++j) {
            acc1[j] = __builtin_amdgcn_wmma_f32_16x16x32_bf16(
                false, af, false, f1[j], (short)0, acc1[j], false, false);
            acc3[j] = __builtin_amdgcn_wmma_f32_16x16x32_bf16(
                false, af, false, f3[j], (short)0, acc3[j], false, false);
        }
        __syncthreads();
    }

#pragma unroll
    for (int j = 0; j < 4; ++j) {
        int col = n0 + wn * 64 + j * 16 + lrow;
#pragma unroll
        for (int r = 0; r < 8; ++r) {
            int lm = wm * 16 + r + lhalf * 8;
            int code = s_code[lm];
            if (code >= 0) {
                float h1 = acc1[j][r], h3 = acc3[j][r];
                float h = (h1 / (1.0f + __expf(-h1))) * h3;
                Hout[(long)code * INTERN + col] = f2bf(h);
            }
        }
    }
}

// ---------------------------------------------------------------------------
// Pass 2: Out[code] = H @ B. H bf16; B bf16 col-major [N][K]; Out fp32.
// ---------------------------------------------------------------------------
__launch_bounds__(256)
__global__ void moe_gemm_out(const unsigned short* __restrict__ Hin,
                             const unsigned short* __restrict__ Bt,
                             float* __restrict__ Out,
                             const int* __restrict__ rows_all,
                             const int* __restrict__ cnt,
                             int fixedM, long estride) {
    __shared__ __align__(16) unsigned short As[2][64 * 40];
    __shared__ __align__(16) unsigned short Bs[2][128 * 40];
    __shared__ int s_code[64];
    __shared__ int s_arow[64];

    int bx = blockIdx.x;
    int nt = bx & 7;
    int mt = (bx >> 3) & 31;
    int e  = bx >> 8;
    int M  = cnt ? cnt[e] : fixedM;
    if (mt * 64 >= M) return;

    const int* rows = rows_all ? (rows_all + (long)e * T_TOK) : nullptr;
    const unsigned short* B = Bt + (long)e * estride;

    int tid = threadIdx.x;
    if (tid < 64) {
        int gr = mt * 64 + tid;
        int code = (gr < M) ? (rows ? rows[gr] : gr) : -1;
        s_code[tid] = code;
        s_arow[tid] = (code >= 0) ? code : 0;
    }
    __syncthreads();

    int lane = tid & 31, wave = tid >> 5;
    int wm = wave & 3, wn = wave >> 2;
    int lrow = lane & 15, lhalf = lane >> 4;
    int n0 = nt * 128;
    int a_row = tid >> 2, a_seg = tid & 3;

    auto stage = [&](int k0, int buf) {
        async_b128(lds_off32(&As[buf][a_row * 40 + a_seg * 8]),
                   Hin + (long)s_arow[a_row] * INTERN + k0 + a_seg * 8);
#pragma unroll
        for (int i = 0; i < 2; ++i) {
            int q = tid + i * 256;
            int col = q >> 2, seg = q & 3;
            async_b128(lds_off32(&Bs[buf][col * 40 + seg * 8]),
                       B + (long)(n0 + col) * INTERN + k0 + seg * 8);
        }
    };

    v8f zero8 = {0.f, 0.f, 0.f, 0.f, 0.f, 0.f, 0.f, 0.f};
    v8f acc[4];
#pragma unroll
    for (int j = 0; j < 4; ++j) acc[j] = zero8;

    stage(0, 0);
    for (int ks = 0; ks < INTERN / 32; ++ks) {
        int cur = ks & 1;
        if (ks + 1 < INTERN / 32) {
            stage((ks + 1) * 32, cur ^ 1);
            WAIT_ASYNC(0x3);
        } else {
            WAIT_ASYNC(0x0);
        }
        __syncthreads();

        const unsigned short* ap = &As[cur][(wm * 16 + lrow) * 40 + lhalf * 16];
        v16bf af = load_frag(ap, ap + 8);
        v16bf fb[4];
#pragma unroll
        for (int j = 0; j < 4; ++j) {
            const unsigned short* bp = &Bs[cur][(wn * 64 + j * 16 + lrow) * 40 + lhalf * 16];
            fb[j] = load_frag(bp, bp + 8);
        }
        __builtin_amdgcn_sched_barrier(0);
#pragma unroll
        for (int j = 0; j < 4; ++j)
            acc[j] = __builtin_amdgcn_wmma_f32_16x16x32_bf16(
                false, af, false, fb[j], (short)0, acc[j], false, false);
        __syncthreads();
    }

#pragma unroll
    for (int j = 0; j < 4; ++j) {
        int col = n0 + wn * 64 + j * 16 + lrow;
#pragma unroll
        for (int r = 0; r < 8; ++r) {
            int lm = wm * 16 + r + lhalf * 8;
            int code = s_code[lm];
            if (code >= 0) Out[(long)code * DIM + col] = acc[j][r];
        }
    }
}

// ---------------------------------------------------------------------------
__launch_bounds__(256)
__global__ void moe_combine(float* __restrict__ out,
                            const float* __restrict__ eo,
                            const float* __restrict__ top_wt) {
    long idx = (long)blockIdx.x * 256 + threadIdx.x;
    int t = (int)(idx >> 10);
    int d = (int)(idx & 1023);
    float v = out[idx];
    v += top_wt[t * 2 + 0] * eo[((long)t * 2 + 0) * DIM + d];
    v += top_wt[t * 2 + 1] * eo[((long)t * 2 + 1) * DIM + d];
    out[idx] = v;
}

// ---------------------------------------------------------------------------
extern "C" void kernel_launch(void* const* d_in, const int* in_sizes, int n_in,
                              void* d_out, int out_size, void* d_ws, size_t ws_size,
                              hipStream_t stream) {
    (void)in_sizes; (void)n_in; (void)out_size; (void)ws_size;
    const float* x   = (const float*)d_in[0];
    const float* gw  = (const float*)d_in[1];
    const float* w1  = (const float*)d_in[2];
    const float* w2  = (const float*)d_in[3];
    const float* w3  = (const float*)d_in[4];
    const float* sw1 = (const float*)d_in[5];
    const float* sw2 = (const float*)d_in[6];
    const float* sw3 = (const float*)d_in[7];
    float* out = (float*)d_out;

    char* ws = (char*)d_ws;
    int*            cnt      = (int*)(ws + 0);
    float*          twt      = (float*)(ws + 4096);
    int*            lists    = (int*)(ws + 20480);
    unsigned short* xb       = (unsigned short*)(ws + 131072);     // 4 MB
    unsigned short* h_shared = (unsigned short*)(ws + 4325376);    // 4 MB
    unsigned short* h_routed = (unsigned short*)(ws + 8519680);    // 8 MB
    float*          eo       = (float*)(ws + 16908288);            // 16 MB
    unsigned short* w1b      = (unsigned short*)(ws + 33685504);   // 16 MB
    unsigned short* w3b      = (unsigned short*)(ws + 50462720);   // 16 MB
    unsigned short* w2b      = (unsigned short*)(ws + 67239936);   // 16 MB
    unsigned short* sw1b     = (unsigned short*)(ws + 84017152);   // 2 MB
    unsigned short* sw3b     = (unsigned short*)(ws + 86114304);   // 2 MB
    unsigned short* sw2b     = (unsigned short*)(ws + 88211456);   // 2 MB

    hipMemsetAsync(cnt, 0, NE * sizeof(int), stream);

    // One-time conversions: x -> bf16; weights -> bf16 B^T ([N][K]).
    convert_x<<<(T_TOK * DIM) / 4 / 256, 256, 0, stream>>>(x, xb);
    dim3 tg(32, 32, NE), ts(32, 32, 1);
    convert_transpose<<<tg, 256, 0, stream>>>(w1, w1b);
    convert_transpose<<<tg, 256, 0, stream>>>(w3, w3b);
    convert_transpose<<<tg, 256, 0, stream>>>(w2, w2b);
    convert_transpose<<<ts, 256, 0, stream>>>(sw1, sw1b);
    convert_transpose<<<ts, 256, 0, stream>>>(sw3, sw3b);
    convert_transpose<<<ts, 256, 0, stream>>>(sw2, sw2b);

    moe_router<<<T_TOK / 256, 256, 0, stream>>>(x, gw, twt, cnt, lists);

    moe_gemm_swiglu<<<NE * 32 * 8, 256, 0, stream>>>(
        xb, w1b, w3b, h_routed, lists, cnt, 0, /*shiftA=*/1, (long)DIM * INTERN);
    moe_gemm_swiglu<<<1 * 32 * 8, 256, 0, stream>>>(
        xb, sw1b, sw3b, h_shared, nullptr, nullptr, T_TOK, /*shiftA=*/0, 0);

    moe_gemm_out<<<NE * 32 * 8, 256, 0, stream>>>(
        h_routed, w2b, eo, lists, cnt, 0, (long)INTERN * DIM);
    moe_gemm_out<<<1 * 32 * 8, 256, 0, stream>>>(
        h_shared, sw2b, out, nullptr, nullptr, T_TOK, 0);

    moe_combine<<<(T_TOK * DIM) / 256, 256, 0, stream>>>(out, eo, twt);
}
